// NCESoftmaxLoss_24361054503573
// MI455X (gfx1250) — matrix operand order repaired and checked by reading, not compiled, not run on hardware
//
#include <hip/hip_runtime.h>
#include <hip/hip_bf16.h>
#include <math.h>

// ---------------------------------------------------------------------------
// NT-Xent (NCE softmax) loss, fused GEMM + shifted-sum LSE, CDNA5 (gfx1250).
//   B=4096, D=512, N=2B=8192, T=0.07
//   sim = normalize(feats) @ normalize(feats)^T / T      (never materialized)
//   loss = mean_i( LSE_{j!=i}(sim[i,j]) - sim[i,(i+B)%N] )
//
// Key numeric insight: rows are L2-normalized => sim in [-1,1] =>
// sim/T in [-C, C] with C = 1/T = 14.2857. Fixed-shift exp(sim/T - C) is in
// [3.9e-13, 1]: no overflow/underflow, so no online max is needed:
//   lse = C + log( sum_{j!=i} exp(sim/T - C) )
// Epilogue per element: 1 fma + 1 exp + 1 cndmask + 1 add (vs 2 exp + max).
//
// Precision: f16 features, fp32 WMMA accumulation (V_WMMA_F32_16X16X32_F16).
// Roofline: 68.7 GFLOP matrix work; HBM ~25MB (~1us @ 23.3TB/s); f16 feats
// (8MB) L2-resident; per-WG B loads identical across waves -> WGP$ broadcast;
// WGP-scope prefetch warms L0 for the next j-tile during current WMMAs.
// ---------------------------------------------------------------------------

typedef _Float16 hv16 __attribute__((ext_vector_type(16)));
typedef _Float16 hv8  __attribute__((ext_vector_type(8)));
typedef float    fv8  __attribute__((ext_vector_type(8)));
typedef float    fv4  __attribute__((ext_vector_type(4)));

#define NROWS   8192
#define DDIM    512
#define NTILES  512                       // 8192 / 16
#define INV_T   14.285714285714286f      // 1/0.07 == fixed LSE shift C

// ---------------------------------------------------------------------------
// Kernel 1: L2-normalize rows of concat(logits, label) -> f16 feats (row-major)
// One wave per row; lane handles 16 contiguous elements.
// ---------------------------------------------------------------------------
__global__ __launch_bounds__(256) void nrm_kernel(
    const float* __restrict__ logits, const float* __restrict__ label,
    _Float16* __restrict__ feats)
{
  int wid  = (blockIdx.x * blockDim.x + threadIdx.x) >> 5;   // row 0..8191
  int lane = threadIdx.x & 31;
  const float* src = (wid < 4096) ? (logits + wid * DDIM)
                                  : (label + (wid - 4096) * DDIM);
  const fv4* s4 = (const fv4*)(src + lane * 16);
  fv4 x0 = s4[0], x1 = s4[1], x2 = s4[2], x3 = s4[3];
  float ss = 0.f;
#pragma unroll
  for (int z = 0; z < 4; ++z)
    ss += x0[z]*x0[z] + x1[z]*x1[z] + x2[z]*x2[z] + x3[z]*x3[z];
#pragma unroll
  for (int m = 1; m < 32; m <<= 1) ss += __shfl_xor(ss, m, 32);
  float scale = rsqrtf(fmaxf(ss, 1e-24f));   // norm clamped like eps=1e-12

  hv8 o0, o1;
#pragma unroll
  for (int z = 0; z < 4; ++z) {
    o0[z]     = (_Float16)(x0[z] * scale);
    o0[z + 4] = (_Float16)(x1[z] * scale);
    o1[z]     = (_Float16)(x2[z] * scale);
    o1[z + 4] = (_Float16)(x3[z] * scale);
  }
  hv8* dst = (hv8*)(feats + wid * DDIM + lane * 16);
  dst[0] = o0; dst[1] = o1;
}

// ---------------------------------------------------------------------------
// Kernel 2: fused sim-tile GEMM (WMMA f16 -> f32) + shifted exp-sum + pos.
// grid = (64 i-blocks, 8 j-chunks), block = 256 (8 waves).
// Wave w owns i-tile it = blockIdx.x*8 + w (16 rows); all waves sweep the
// SAME 64 j-tiles of chunk jc (B fragments shared across waves via WGP$).
// A fragments (16 x v16h = 128 VGPR) live in registers for the whole sweep.
// Per-lane state: 8 running exp-sums + 8 pos slots; summed across the
// 16-lane half with shfl_xor; (s, pos) per row -> workspace.
// ---------------------------------------------------------------------------
__global__ __launch_bounds__(256, 1) void simlse_kernel(
    const _Float16* __restrict__ feats, float2* __restrict__ partial)
{
  const int lane  = threadIdx.x & 31;
  const int wave  = threadIdx.x >> 5;
  const int n     = lane & 15;        // column within tile (C/D layout)
  const int hi    = lane >> 4;        // half-wave
  const int rbase = hi << 3;          // C rows: hi ? 8..15 : 0..7
  const int it    = blockIdx.x * 8 + wave;
  const int jc    = blockIdx.y;
  const int i0    = it << 4;
  const int jt_pos = (it + 256) & (NTILES - 1);   // (i + 4096)/16 mod 512

  // --- preload A fragments (ISA A-layout: lane m=n, K split 0..7/16..23 per half)
  hv16 a[16];
  {
    const _Float16* arow = feats + (i0 + n) * DDIM + (hi << 3);
#pragma unroll
    for (int ks = 0; ks < 16; ++ks) {
      hv8 lo = *(const hv8*)(arow + ks * 32);
      hv8 hh = *(const hv8*)(arow + ks * 32 + 16);
      a[ks] = __builtin_shufflevector(lo, hh, 0, 1, 2, 3, 4, 5, 6, 7,
                                              8, 9, 10, 11, 12, 13, 14, 15);
    }
  }

  float s8[8], p8[8];
#pragma unroll
  for (int r = 0; r < 8; ++r) { s8[r] = 0.f; p8[r] = 0.f; }

  for (int t = 0; t < 64; ++t) {
    const int jt = (jc << 6) + t;
    const int j0 = jt << 4;
    // B = N^T tile: ISA B-layout => lane n holds K 8h..8h+7, 16+8h..16+8h+7 of row j0+n.
    const _Float16* brow = feats + (j0 + n) * DDIM + (hi << 3);
    if (t < 63)  // WGP-scope prefetch: warm WGP$/L0 with next j-tile's lines
      __builtin_prefetch(brow + 16 * DDIM, 0, 3);

    fv8 acc = {0.f, 0.f, 0.f, 0.f, 0.f, 0.f, 0.f, 0.f};
#pragma unroll
    for (int ks = 0; ks < 16; ++ks) {
      hv16 b = __builtin_shufflevector(*(const hv8*)(brow + (ks << 5)), *(const hv8*)(brow + (ks << 5) + 16), 0, 1, 2, 3, 4, 5, 6, 7, 8, 9, 10, 11, 12, 13, 14, 15);
      acc = __builtin_amdgcn_wmma_f32_16x16x32_f16(
          /*neg_a=*/false, a[ks], /*neg_b=*/false, b,
          /*c_mod=*/(short)0, acc, /*reuse_a=*/false, /*reuse_b=*/false);
    }

    const bool diagT = (jt == it);
    const bool posT  = (jt == jt_pos);
#pragma unroll
    for (int r = 0; r < 8; ++r) {
      const bool hit = (n == rbase + r);           // tile-diagonal element
      // e = exp(sim/T - C) = exp((acc-1)*C); argument in [-2C, 0] -> exact
      float e = __expf(fmaf(acc[r], INV_T, -INV_T));
      if (diagT && hit) e = 0.f;                   // mask self-similarity
      s8[r] += e;
      if (posT && hit) p8[r] += acc[r] * INV_T;    // positive-pair logit
    }
  }

  // --- sum (s, pos) across the 16 lanes of each half-wave
#pragma unroll
  for (int r = 0; r < 8; ++r) {
    float s = s8[r], p = p8[r];
#pragma unroll
    for (int off = 1; off < 16; off <<= 1) {
      s += __shfl_xor(s, off, 32);
      p += __shfl_xor(p, off, 32);
    }
    s8[r] = s; p8[r] = p;
  }

  if (n == 0) {  // lanes 0 (rows i0..i0+7) and 16 (rows i0+8..i0+15)
#pragma unroll
    for (int r = 0; r < 8; ++r) {
      int row = i0 + rbase + r;
      partial[(jc << 13) + row] = make_float2(s8[r], p8[r]);
    }
  }
}

// ---------------------------------------------------------------------------
// Kernel 3: per-row sum of the 8 j-chunk partials (fixed order), rowloss =
// (C + log S) - pos, then fixed-order block tree reduction -> blockSums[32].
// Deterministic: no atomics, fixed order.
// ---------------------------------------------------------------------------
__global__ __launch_bounds__(256) void rowmerge_kernel(
    const float2* __restrict__ partial, float* __restrict__ blockSums)
{
  int row = blockIdx.x * 256 + threadIdx.x;   // 32 blocks x 256 = 8192
  float S = 0.f, P = 0.f;
#pragma unroll
  for (int c = 0; c < 8; ++c) {
    float2 w = partial[(c << 13) + row];
    S += w.x;
    P += w.y;
  }
  float rowloss = (INV_T + logf(S)) - P;

  __shared__ float red[256];
  red[threadIdx.x] = rowloss;
  __syncthreads();
#pragma unroll
  for (int s = 128; s > 0; s >>= 1) {
    if (threadIdx.x < s) red[threadIdx.x] += red[threadIdx.x + s];
    __syncthreads();
  }
  if (threadIdx.x == 0) blockSums[blockIdx.x] = red[0];
}

__global__ void final_kernel(const float* __restrict__ blockSums,
                             float* __restrict__ out)
{
  if (threadIdx.x == 0) {
    float t = 0.f;
    for (int i = 0; i < 32; ++i) t += blockSums[i];
    out[0] = t * (1.0f / 8192.0f);
  }
}

// ---------------------------------------------------------------------------
extern "C" void kernel_launch(void* const* d_in, const int* in_sizes, int n_in,
                              void* d_out, int out_size, void* d_ws, size_t ws_size,
                              hipStream_t stream) {
  const float* logits = (const float*)d_in[0];
  const float* label  = (const float*)d_in[1];

  // workspace layout
  _Float16* feats     = (_Float16*)d_ws;                                   // 8 MB
  float2*   partial   = (float2*)((char*)d_ws + (size_t)8 * 1024 * 1024);  // 512 KB
  float*    blockSums = (float*)((char*)d_ws + (size_t)8 * 1024 * 1024
                                             + (size_t)512 * 1024);        // 128 B

  nrm_kernel<<<NROWS / 8, 256, 0, stream>>>(logits, label, feats);
  simlse_kernel<<<dim3(64, 8), 256, 0, stream>>>(feats, partial);
  rowmerge_kernel<<<32, 256, 0, stream>>>(partial, blockSums);
  final_kernel<<<1, 32, 0, stream>>>(blockSums, (float*)d_out);
}
